// Top2Router_79413945303481
// MI455X (gfx1250) — matrix operand order repaired and checked by reading, not compiled, not run on hardware
//
#include <hip/hip_runtime.h>

#define E 8
#define ROUTER_TPB 1024
#define NWAVES (ROUTER_TPB / 32)

typedef _Float16 v16h __attribute__((ext_vector_type(16)));
typedef float    v8f  __attribute__((ext_vector_type(8)));
typedef float    f4   __attribute__((ext_vector_type(4)));
typedef int      v4i  __attribute__((ext_vector_type(4)));

#if __has_builtin(__builtin_amdgcn_global_load_async_to_lds_b128) && \
    __has_builtin(__builtin_amdgcn_s_wait_asynccnt)
#define HAVE_ASYNC_LDS 1
#else
#define HAVE_ASYNC_LDS 0
#endif

typedef __attribute__((address_space(1))) v4i* gvec_t;   // global int4*
typedef __attribute__((address_space(3))) v4i* lvec_t;   // LDS int4*

// ---------------------------------------------------------------------------
// Kernel 1: zero-fill the 1.34 GB output with nontemporal b128 stores.
// Pure store-bandwidth bound: ~58 us floor at 23.3 TB/s HBM.
// ---------------------------------------------------------------------------
__global__ void fill_zero_kernel(float* __restrict__ out, long long n) {
  const long long n4     = n >> 2;
  const long long stride = (long long)gridDim.x * blockDim.x;
  long long       i      = (long long)blockIdx.x * blockDim.x + threadIdx.x;
  f4 z = {0.0f, 0.0f, 0.0f, 0.0f};
  f4* o4 = (f4*)out;
  for (long long j = i; j < n4; j += stride)
    __builtin_nontemporal_store(z, o4 + j);
  if (i == 0)
    for (long long j = (n4 << 2); j < n; ++j) out[j] = 0.0f;
}

// ---------------------------------------------------------------------------
// Kernel 2: single-workgroup top-2 router.
// 1024 threads (32 wave32 waves), thread t owns tokens [8t, 8t+8).
//   - logits staged global->LDS with async-to-LDS copies (ASYNCcnt)
//   - intra-wave 32x16 ordered prefix = triangular 32x32 matmul (2x WMMA f16)
//   - cross-wave carry: tiny serial scan by 16 threads
//   - softmax weights scattered into the pre-zeroed output
// ---------------------------------------------------------------------------
__global__ __launch_bounds__(ROUTER_TPB) void top2_router_kernel(
    const float* __restrict__ in, float* __restrict__ out, int s, int cap) {
  const int tid  = threadIdx.x;
  const int lane = tid & 31;
  const int wave = tid >> 5;
  const int hi   = lane >> 4;   // which half of the wave (K split)
  const int lm   = lane & 15;   // row/col within 16-wide tile

  // dynamic LDS layout: [ logits s*E f32 | xch | waveExcl | totals ]
  extern __shared__ char smem[];
  float* lg = (float*)smem;
  const size_t lgBytes = (size_t)s * E * sizeof(float);
  _Float16 (*xch)[32][16] = (_Float16(*)[32][16])(smem + lgBytes);
  int (*waveExcl)[16]     = (int(*)[16])(smem + lgBytes + (size_t)NWAVES * 32 * 16 * 2);
  int* totals             = (int*)(smem + lgBytes + (size_t)NWAVES * 32 * 16 * 2 +
                                   (size_t)NWAVES * 16 * 4);

  const int tokPer = (s + ROUTER_TPB - 1) / ROUTER_TPB;
  const int t0     = tid * tokPer;
  const int nTok   = (t0 >= s) ? 0 : ((s - t0 < tokPer) ? (s - t0) : tokPer);

  // ---- stage this thread's token rows into LDS (async, no barrier needed:
  //      each thread only consumes the rows it staged itself) ----
  {
    const float* src = in + (long long)t0 * E;
    float*       dst = lg + (long long)t0 * E;
    const int bytes  = nTok * E * (int)sizeof(float);  // multiple of 16 (E=8)
    if (bytes > 0) __builtin_prefetch(src, 0, 1);      // global_prefetch_b8
#if HAVE_ASYNC_LDS
    for (int b = 0; b < bytes; b += 16)
      __builtin_amdgcn_global_load_async_to_lds_b128(
          (gvec_t)((const char*)src + b), (lvec_t)((char*)dst + b), 0, 0);
    __builtin_amdgcn_s_wait_asynccnt(0);
#else
    for (int f = 0; f < nTok * E; f += 4)
      *(f4*)(dst + f) = *(const f4*)(src + f);
#endif
  }

  // ---- pass 1: per-thread counters (argmax needs no softmax) ----
  int cnt[16];
#pragma unroll
  for (int n = 0; n < 16; ++n) cnt[n] = 0;

  for (int k = 0; k < nTok; ++k) {
    const float* px = lg + (long long)(t0 + k) * E;
    float x[E];
#pragma unroll
    for (int j = 0; j < E; ++j) x[j] = px[j];
    int a1 = 0; float b1 = x[0];
#pragma unroll
    for (int j = 1; j < E; ++j) if (x[j] > b1) { b1 = x[j]; a1 = j; }
    int a2 = -1; float b2 = -3.0e38f;
#pragma unroll
    for (int j = 0; j < E; ++j) {
      if (j == a1) continue;
      if (x[j] > b2) { b2 = x[j]; a2 = j; }
    }
    cnt[a1] += 1;
    cnt[8 + a2] += 1;
  }

  // publish counts: xch[wave][K=lane][N=counter], exact in f16 (<= 8)
#pragma unroll
  for (int n = 0; n < 16; ++n) xch[wave][lane][n] = (_Float16)cnt[n];
  __syncthreads();

  // ---- intra-wave inclusive scan via two WMMAs: D = T(32x32) @ M(32x16) ----
  // B operand (32x16, f16): this lane holds column N=lm, K = 16*hi + h.
  v16h bm;
#pragma unroll
  for (int h = 0; h < 16; ++h) bm[h] = xch[wave][16 * hi + h][lm];

  // A operands (16x32, f16), row M = lm, K(lane,h) matching B:
  //   K = 16*hi + h
  v16h aTop, aBot;
#pragma unroll
  for (int h = 0; h < 16; ++h) {
    const int K = 16 * hi + h;
    // [ T16 | 0 ]  -> rows 0..15 of the 32-row prefix
    aTop[h] = (K < 16 && K <= lm) ? (_Float16)1.0f : (_Float16)0.0f;
    // [ ones | T16 ] -> rows 16..31
    aBot[h] = (K < 16 || (K - 16) <= lm) ? (_Float16)1.0f : (_Float16)0.0f;
  }

  v8f cz = {};
  v8f dTop = __builtin_amdgcn_wmma_f32_16x16x32_f16(
      false, aTop, false, bm, (short)0, cz, false, false);
  v8f dBot = __builtin_amdgcn_wmma_f32_16x16x32_f16(
      false, aBot, false, bm, (short)0, cz, false, false);

  // store inclusive rows back (exact: sums <= 256): row = v + 8*hi (+16 bot)
#pragma unroll
  for (int v = 0; v < 8; ++v) xch[wave][v + 8 * hi][lm]      = (_Float16)dTop[v];
#pragma unroll
  for (int v = 0; v < 8; ++v) xch[wave][16 + v + 8 * hi][lm] = (_Float16)dBot[v];
  __syncthreads();

  // ---- cross-wave exclusive carry scan (16 counters, 32 waves) ----
  if (tid < 16) {
    int run = 0;
    for (int w = 0; w < NWAVES; ++w) {
      waveExcl[w][tid] = run;
      run += (int)(float)xch[w][31][tid];
    }
    totals[tid] = run;  // counters 0..7 = total top1 count per expert
  }
  __syncthreads();

  // this thread's exclusive base per counter
  int slot[16];
#pragma unroll
  for (int n = 0; n < 16; ++n)
    slot[n] = waveExcl[wave][n] + (int)(float)xch[wave][lane][n] - cnt[n];

  // ---- pass 2: recompute per-token routing from LDS, softmax, scatter ----
  const long long maskBase = (long long)s * E * cap;
  for (int k = 0; k < nTok; ++k) {
    const int i = t0 + k;
    const float* px = lg + (long long)i * E;
    float x[E];
#pragma unroll
    for (int j = 0; j < E; ++j) x[j] = px[j];

    int a1 = 0; float b1 = x[0];
#pragma unroll
    for (int j = 1; j < E; ++j) if (x[j] > b1) { b1 = x[j]; a1 = j; }
    int a2 = -1; float b2 = -3.0e38f;
#pragma unroll
    for (int j = 0; j < E; ++j) {
      if (j == a1) continue;
      if (x[j] > b2) { b2 = x[j]; a2 = j; }
    }

    float m = x[0];
#pragma unroll
    for (int j = 1; j < E; ++j) m = fmaxf(m, x[j]);
    float ex[E]; float sum = 0.0f;
#pragma unroll
    for (int j = 0; j < E; ++j) { ex[j] = __expf(x[j] - m); sum += ex[j]; }
    const float inv = 1.0f / sum;
    const float w1 = ex[a1] * inv;
    const float w2 = ex[a2] * inv;

    const int r1 = slot[a1];                  slot[a1]++;
    const int r2 = totals[a2] + slot[8 + a2]; slot[8 + a2]++;

    if (r1 < cap) {
      const long long o = ((long long)i * E + a1) * cap + r1;
      out[o]            = w1;
      out[maskBase + o] = 1.0f;
    }
    if (r2 < cap) {
      const long long o = ((long long)i * E + a2) * cap + r2;
      out[o]            = w2;
      out[maskBase + o] = 1.0f;
    }
  }
}

// ---------------------------------------------------------------------------
extern "C" void kernel_launch(void* const* d_in, const int* in_sizes, int n_in,
                              void* d_out, int out_size, void* d_ws, size_t ws_size,
                              hipStream_t stream) {
  const float* in  = (const float*)d_in[0];
  float*       out = (float*)d_out;

  const int s = in_sizes[0] / E;  // 8192 tokens
  // capacity = floor(2 * 1.25 * s / e); round up to even; min 4
  int cap = (int)((2.5 * (double)s) / (double)E);
  cap += (cap & 1);
  if (cap < 4) cap = 4;

  const long long n = (long long)out_size;  // cb_weight + sec_mask, flat

  // dynamic LDS: logits + scan exchange buffers (~290 KB of the 320 KB WGP LDS)
  const size_t smemBytes = (size_t)s * E * sizeof(float) +
                           (size_t)NWAVES * 32 * 16 * 2 +
                           (size_t)NWAVES * 16 * 4 + 16 * 4;

  fill_zero_kernel<<<8192, 256, 0, stream>>>(out, n);
  top2_router_kernel<<<1, ROUTER_TPB, smemBytes, stream>>>(in, out, s, cap);
}
